// TinyBatched_17386027615043
// MI455X (gfx1250) — matrix operand order repaired and checked by reading, not compiled, not run on hardware
//
#include <hip/hip_runtime.h>

// Problem constants (from the reference)
#define D_IN  512
#define NTOT  32896     // 257 * 128
#define BATCH 4096

// Fallback kernel tile
constexpr int BM = 64;
constexpr int BN = 128;
constexpr int BK = 32;
constexpr int NK = D_IN / BK; // 16 K-steps
constexpr int ASTR = 40;      // padded LDS row stride (bf16 units)
constexpr int BSTR = 40;

// Fast (async) kernel tile: 128x128 block, 8 waves = 4(M) x 2(N),
// each wave owns 2 Mtiles x 4 Ntiles = 24 WMMAs per K-step.
constexpr int GBM = 128;
constexpr int GBN = 128;

typedef __attribute__((ext_vector_type(16))) __bf16 v16bf;
typedef __attribute__((ext_vector_type(8)))  float  v8f;

union Frag32B { uint4 q[2]; v16bf v; };

__device__ __forceinline__ unsigned f_as_u(float f) {
    union { float f; unsigned u; } c; c.f = f; return c.u;
}
__device__ __forceinline__ float u_as_f(unsigned u) {
    union { unsigned u; float f; } c; c.u = u; return c.f;
}
// round-to-nearest-even fp32 -> bf16 (raw u16)
__device__ __forceinline__ unsigned short f2bf(float f) {
    unsigned u = f_as_u(f);
    u += 0x7FFFu + ((u >> 16) & 1u);
    return (unsigned short)(u >> 16);
}
__device__ __forceinline__ float bf2f(unsigned short h) {
    return u_as_f(((unsigned)h) << 16);
}
__device__ __forceinline__ void cvt4(float4 f, uint2& hi, uint2& lo) {
    unsigned short h0 = f2bf(f.x), h1 = f2bf(f.y), h2 = f2bf(f.z), h3 = f2bf(f.w);
    unsigned short l0 = f2bf(f.x - bf2f(h0));
    unsigned short l1 = f2bf(f.y - bf2f(h1));
    unsigned short l2 = f2bf(f.z - bf2f(h2));
    unsigned short l3 = f2bf(f.w - bf2f(h3));
    hi.x = (unsigned)h0 | ((unsigned)h1 << 16);
    hi.y = (unsigned)h2 | ((unsigned)h3 << 16);
    lo.x = (unsigned)l0 | ((unsigned)l1 << 16);
    lo.y = (unsigned)l2 | ((unsigned)l3 << 16);
}

// ---- CDNA5 async global->LDS copy (ASYNCcnt-tracked), 16B per lane ----
__device__ __forceinline__ void async_b128(unsigned lds_byte_off, const void* gptr) {
    asm volatile("global_load_async_to_lds_b128 %0, %1, off"
                 :: "v"(lds_byte_off), "v"(gptr) : "memory");
}
__device__ __forceinline__ void wait_async0() {
    asm volatile("s_wait_asynccnt 0x0" ::: "memory");
}

// =====================================================================
// Pre-pass 1: x fp32 -> xh/xl bf16 [BATCH][D_IN]  (streaming)
// =====================================================================
__global__ __launch_bounds__(256)
void convert_x(const float* __restrict__ x,
               unsigned short* __restrict__ xh,
               unsigned short* __restrict__ xl)
{
    const size_t i = (size_t)blockIdx.x * blockDim.x + threadIdx.x;  // float4 idx
    float4 f = ((const float4*)x)[i];
    uint2 h, l;
    cvt4(f, h, l);
    ((uint2*)xh)[i] = h;
    ((uint2*)xl)[i] = l;
}

// =====================================================================
// Pre-pass 2: W fp32 [D_IN][NTOT] -> wh/wl bf16 TRANSPOSED [NTOT][D_IN]
// LDS-tiled 32(k) x 64(n) transpose per block.
// =====================================================================
__global__ __launch_bounds__(256)
void convert_w(const float* __restrict__ W,
               unsigned short* __restrict__ wh,
               unsigned short* __restrict__ wl)
{
    __shared__ unsigned short Th[64 * 40];   // [n][k], padded stride 40
    __shared__ unsigned short Tl[64 * 40];

    const int tid = threadIdx.x;
    const int n0  = blockIdx.x * 64;
    const int k0  = blockIdx.y * 32;

    const int k  = tid >> 3;          // 0..31
    const int nq = (tid & 7) * 8;     // 0..56
#pragma unroll
    for (int j = 0; j < 2; ++j) {
        float4 f = *(const float4*)(W + (size_t)(k0 + k) * NTOT + n0 + nq + 4 * j);
        const float fv[4] = { f.x, f.y, f.z, f.w };
#pragma unroll
        for (int e = 0; e < 4; ++e) {
            unsigned short hh = f2bf(fv[e]);
            unsigned short ll = f2bf(fv[e] - bf2f(hh));
            const int n = nq + 4 * j + e;
            Th[n * 40 + k] = hh;
            Tl[n * 40 + k] = ll;
        }
    }
    __syncthreads();
    const int n  = tid >> 2;          // 0..63
    const int ch = (tid & 3) * 8;     // 0..24 (bf16 units)
    uint4 hq = *(const uint4*)&Th[n * 40 + ch];
    uint4 lq = *(const uint4*)&Tl[n * 40 + ch];
    *(uint4*)(wh + (size_t)(n0 + n) * D_IN + k0 + ch) = hq;
    *(uint4*)(wl + (size_t)(n0 + n) * D_IN + k0 + ch) = lq;
}

// =====================================================================
// Main GEMM: 128x128 block tile, bf16x3 fp32 emulation, async
// global->LDS staging, double-buffered LDS, 24 WMMAs per wave per K-step.
// =====================================================================
__global__ __launch_bounds__(256)
void gemm_async_bf16x3(const unsigned short* __restrict__ xh,
                       const unsigned short* __restrict__ xl,
                       const unsigned short* __restrict__ wh,  // [NTOT][D_IN]
                       const unsigned short* __restrict__ wl,
                       const float* __restrict__ bias,
                       float* __restrict__ out)
{
    __shared__ unsigned short Ah[2][GBM * ASTR];   // 2 x 128 x 40 u16 = 20 KB
    __shared__ unsigned short Al[2][GBM * ASTR];
    __shared__ unsigned short Bh[2][GBN * BSTR];
    __shared__ unsigned short Bl[2][GBN * BSTR];   // total 80 KB

    const int tid  = threadIdx.x;
    const int m0   = blockIdx.y * GBM;
    const int n0   = blockIdx.x * GBN;

    const int lane  = tid & 31;
    const int wid   = tid >> 5;
    const int waveM = wid & 3;     // 0..3 -> 32-row strip (2 Mtiles)
    const int waveN = wid >> 2;    // 0..1 -> 64-col strip (4 Ntiles)
    const int half  = lane >> 4;
    const int l16   = lane & 15;

    // LDS base byte offsets (low 32 bits of the generic pointer = LDS offset)
    const unsigned ah0 = (unsigned)(unsigned long long)&Ah[0][0];
    const unsigned al0 = (unsigned)(unsigned long long)&Al[0][0];
    const unsigned bh0 = (unsigned)(unsigned long long)&Bh[0][0];
    const unsigned bl0 = (unsigned)(unsigned long long)&Bl[0][0];

    // async staging map: 128 rows x 4 chunks(16B) per array -> 2 chunks/thread
    const int rr = tid >> 2;             // row (j=0) 0..63; j=1 adds 64
    const int ch = (tid & 3) * 8;        // k offset (bf16 units)

    auto issueStage = [&](int ks, int buf) {
        const int kb = ks * BK;
        const unsigned abuf = (unsigned)(buf * GBM * ASTR * 2);
        const unsigned bbuf = (unsigned)(buf * GBN * BSTR * 2);
#pragma unroll
        for (int j = 0; j < 2; ++j) {
            const int r = rr + 64 * j;
            const unsigned l_off = (unsigned)(r * ASTR + ch) * 2u;
            const size_t   a_g   = (size_t)(m0 + r) * D_IN + kb + ch;
            const size_t   b_g   = (size_t)(n0 + r) * D_IN + kb + ch;
            async_b128(ah0 + abuf + l_off, xh + a_g);
            async_b128(al0 + abuf + l_off, xl + a_g);
            async_b128(bh0 + bbuf + l_off, wh + b_g);
            async_b128(bl0 + bbuf + l_off, wl + b_g);
        }
    };

    v8f acc[2][4];
#pragma unroll
    for (int mt = 0; mt < 2; ++mt)
#pragma unroll
        for (int nt = 0; nt < 4; ++nt) acc[mt][nt] = v8f{};

    auto computeStage = [&](int buf) {
        Frag32B a_hi[2], a_lo[2];
#pragma unroll
        for (int mt = 0; mt < 2; ++mt) {
            const int a_base = (waveM * 32 + mt * 16 + l16) * ASTR + 8 * half;
            const uint4* pAh = (const uint4*)&Ah[buf][a_base];
            const uint4* pAl = (const uint4*)&Al[buf][a_base];
            a_hi[mt].q[0] = pAh[0]; a_hi[mt].q[1] = pAh[2];  // +16 bf16 -> +2 uint4
            a_lo[mt].q[0] = pAl[0]; a_lo[mt].q[1] = pAl[2];
        }
#pragma unroll
        for (int nt = 0; nt < 4; ++nt) {
            const int n = waveN * 64 + nt * 16 + l16;
            Frag32B b_hi, b_lo;
            const uint4* pBh = (const uint4*)&Bh[buf][n * BSTR + 8 * half];
            const uint4* pBl = (const uint4*)&Bl[buf][n * BSTR + 8 * half];
            b_hi.q[0] = pBh[0]; b_hi.q[1] = pBh[2];
            b_lo.q[0] = pBl[0]; b_lo.q[1] = pBl[2];
#pragma unroll
            for (int mt = 0; mt < 2; ++mt) {
                // hi*lo + lo*hi + hi*hi (drop lo*lo ~2^-32)
                acc[mt][nt] = __builtin_amdgcn_wmma_f32_16x16x32_bf16(
                    false, a_hi[mt].v, false, b_lo.v, (short)0, acc[mt][nt], false, false);
                acc[mt][nt] = __builtin_amdgcn_wmma_f32_16x16x32_bf16(
                    false, a_lo[mt].v, false, b_hi.v, (short)0, acc[mt][nt], false, false);
                acc[mt][nt] = __builtin_amdgcn_wmma_f32_16x16x32_bf16(
                    false, a_hi[mt].v, false, b_hi.v, (short)0, acc[mt][nt], false, false);
            }
        }
    };

    issueStage(0, 0);
    wait_async0();
    __syncthreads();
    int cur = 0;
#pragma unroll 1
    for (int ks = 0; ks < NK; ++ks) {
        if (ks + 1 < NK) issueStage(ks + 1, cur ^ 1);   // hide under WMMAs
        computeStage(cur);
        if (ks + 1 < NK) {
            wait_async0();        // drain this wave's async group
            __syncthreads();      // make stage visible to all waves
        }
        cur ^= 1;
    }

    // Epilogue: bias + store per 32-bit C/D layout (lane=N, vgpr row=M, +8 per half)
#pragma unroll
    for (int mt = 0; mt < 2; ++mt) {
        const int crow = m0 + waveM * 32 + mt * 16 + half * 8;
#pragma unroll
        for (int nt = 0; nt < 4; ++nt) {
            const int n = n0 + waveN * 64 + nt * 16 + l16;
            const float bv = bias[n];
#pragma unroll
            for (int r = 0; r < 8; ++r)
                out[(size_t)(crow + r) * NTOT + n] = acc[mt][nt][r] + bv;
        }
    }
}

// =====================================================================
// Fallback (fused, 64x128): used only if d_ws is too small.
// =====================================================================
__global__ __launch_bounds__(256)
void fused_gemm_bias_bf16x3(const float* __restrict__ x,
                            const float* __restrict__ W,
                            const float* __restrict__ bias,
                            float* __restrict__ out)
{
    __shared__ unsigned short Ah[2][BM * ASTR];
    __shared__ unsigned short Al[2][BM * ASTR];
    __shared__ unsigned short Bh[2][BN * BSTR];
    __shared__ unsigned short Bl[2][BN * BSTR];

    const int tid  = threadIdx.x;
    const int m0   = blockIdx.y * BM;
    const int n0   = blockIdx.x * BN;
    const int lane  = tid & 31;
    const int wid   = tid >> 5;
    const int waveM = wid & 3;
    const int waveN = wid >> 2;
    const int half  = lane >> 4;
    const int l16   = lane & 15;
    const int a_base = (waveM * 16 + l16) * ASTR + 8 * half;

    const int am0 = tid >> 3;
    const int am1 = (tid + 256) >> 3;
    const int acq = (tid & 7) * 4;
    const int bk  = tid >> 3;
    const int bn  = (tid & 7) * 16;

    float4 areg[2];
    float4 breg[4];

    auto loadG = [&](int ks) {
        const int kb = ks * BK;
        areg[0] = *(const float4*)(x + (size_t)(m0 + am0) * D_IN + kb + acq);
        areg[1] = *(const float4*)(x + (size_t)(m0 + am1) * D_IN + kb + acq);
#pragma unroll
        for (int i = 0; i < 4; ++i)
            breg[i] = *(const float4*)(W + (size_t)(kb + bk) * NTOT + n0 + bn + 4 * i);
    };

    auto storeL = [&](int buf) {
        uint2 h, l;
        cvt4(areg[0], h, l);
        *(uint2*)&Ah[buf][am0 * ASTR + acq] = h;
        *(uint2*)&Al[buf][am0 * ASTR + acq] = l;
        cvt4(areg[1], h, l);
        *(uint2*)&Ah[buf][am1 * ASTR + acq] = h;
        *(uint2*)&Al[buf][am1 * ASTR + acq] = l;
#pragma unroll
        for (int i = 0; i < 4; ++i) {
            const float fv[4] = { breg[i].x, breg[i].y, breg[i].z, breg[i].w };
#pragma unroll
            for (int e = 0; e < 4; ++e) {
                unsigned short hh = f2bf(fv[e]);
                unsigned short ll = f2bf(fv[e] - bf2f(hh));
                const int n = bn + 4 * i + e;
                Bh[buf][n * BSTR + bk] = hh;
                Bl[buf][n * BSTR + bk] = ll;
            }
        }
    };

    v8f acc[4] = {v8f{}, v8f{}, v8f{}, v8f{}};

    auto computeStage = [&](int buf) {
        Frag32B a_hi, a_lo;
        const uint4* pAh = (const uint4*)&Ah[buf][a_base];
        const uint4* pAl = (const uint4*)&Al[buf][a_base];
        a_hi.q[0] = pAh[0]; a_hi.q[1] = pAh[2];
        a_lo.q[0] = pAl[0]; a_lo.q[1] = pAl[2];
#pragma unroll
        for (int nt = 0; nt < 4; ++nt) {
            const int n = waveN * 64 + nt * 16 + l16;
            Frag32B b_hi, b_lo;
            const uint4* pBh = (const uint4*)&Bh[buf][n * BSTR + 8 * half];
            const uint4* pBl = (const uint4*)&Bl[buf][n * BSTR + 8 * half];
            b_hi.q[0] = pBh[0]; b_hi.q[1] = pBh[2];
            b_lo.q[0] = pBl[0]; b_lo.q[1] = pBl[2];
            acc[nt] = __builtin_amdgcn_wmma_f32_16x16x32_bf16(
                false, a_hi.v, false, b_lo.v, (short)0, acc[nt], false, false);
            acc[nt] = __builtin_amdgcn_wmma_f32_16x16x32_bf16(
                false, a_lo.v, false, b_hi.v, (short)0, acc[nt], false, false);
            acc[nt] = __builtin_amdgcn_wmma_f32_16x16x32_bf16(
                false, a_hi.v, false, b_hi.v, (short)0, acc[nt], false, false);
        }
    };

    loadG(0);
    storeL(0);
    __syncthreads();
    int cur = 0;
#pragma unroll 1
    for (int ks = 0; ks < NK; ++ks) {
        if (ks + 1 < NK) loadG(ks + 1);
        computeStage(cur);
        if (ks + 1 < NK) {
            storeL(cur ^ 1);
            __syncthreads();
        }
        cur ^= 1;
    }

    const int crow = m0 + waveM * 16 + half * 8;
#pragma unroll
    for (int nt = 0; nt < 4; ++nt) {
        const int n = n0 + waveN * 64 + nt * 16 + l16;
        const float bv = bias[n];
#pragma unroll
        for (int r = 0; r < 8; ++r)
            out[(size_t)(crow + r) * NTOT + n] = acc[nt][r] + bv;
    }
}

extern "C" void kernel_launch(void* const* d_in, const int* in_sizes, int n_in,
                              void* d_out, int out_size, void* d_ws, size_t ws_size,
                              hipStream_t stream) {
    const float* x = (const float*)d_in[0];   // [BATCH, D_IN]
    const float* W = (const float*)d_in[1];   // [D_IN, NTOT]
    const float* b = (const float*)d_in[2];   // [NTOT]
    float* out = (float*)d_out;               // [BATCH, NTOT]
    (void)in_sizes; (void)n_in; (void)out_size;

    const size_t wElems = (size_t)NTOT * D_IN;   // 16.8M
    const size_t xElems = (size_t)BATCH * D_IN;  // 2.1M
    const size_t need   = 2 * wElems * 2 + 2 * xElems * 2;  // ~75.8 MB

    dim3 block(256);

    if (ws_size >= need) {
        unsigned short* wh = (unsigned short*)d_ws;
        unsigned short* wl = wh + wElems;
        unsigned short* xh = wl + wElems;
        unsigned short* xl = xh + xElems;
        convert_x<<<(unsigned)(xElems / 4 / 256), block, 0, stream>>>(x, xh, xl);
        convert_w<<<dim3(NTOT / 64, D_IN / 32), block, 0, stream>>>(W, wh, wl);
        dim3 ggrid(NTOT / GBN, BATCH / GBM);     // (257, 32)
        gemm_async_bf16x3<<<ggrid, block, 0, stream>>>(xh, xl, wh, wl, b, out);
    } else {
        dim3 fgrid(NTOT / BN, BATCH / BM);       // (257, 64)
        fused_gemm_bias_bf16x3<<<fgrid, block, 0, stream>>>(x, W, b, out);
    }
}
